// RNN_19396072309352
// MI455X (gfx1250) — matrix operand order repaired and checked by reading, not compiled, not run on hardware
//
#include <hip/hip_runtime.h>

typedef __attribute__((ext_vector_type(16))) _Float16 v16h;
typedef __attribute__((ext_vector_type(8)))  float    v8f;
typedef __attribute__((ext_vector_type(4)))  float    v4f;

#define BATCH 65536
#define SEQ   64
#define IN_F  16
#define H_SZ  10
#define OUT_F 5

#define WAVES_PER_BLOCK 8
#define ROWS_PER_WAVE   16   // one 16x16 WMMA tile of batch rows per wave

__device__ __forceinline__ float fast_tanh(float v) {
#if __has_builtin(__builtin_amdgcn_tanhf)
    return __builtin_amdgcn_tanhf(v);
#else
    // tanh(v) = 1 - 2/(e^{2v}+1)  -> v_exp_f32 + v_rcp_f32
    float e = __expf(2.0f * v);
    return 1.0f - 2.0f * __builtin_amdgcn_rcpf(e + 1.0f);
#endif
}

__device__ __forceinline__ float fast_sigmoid(float v) {
    return __builtin_amdgcn_rcpf(1.0f + __expf(-v));
}

__global__ __launch_bounds__(256) void rnn_scan_wmma(
    const float* __restrict__ x,   // [BATCH, SEQ, IN_F]
    const float* __restrict__ W1,  // [IN_F, H_SZ]
    const float* __restrict__ b1,  // [H_SZ]
    const float* __restrict__ W2,  // [H_SZ, OUT_F]
    const float* __restrict__ b2,  // [OUT_F]
    float* __restrict__ out)       // [BATCH, OUT_F]
{
    const int lane = threadIdx.x & 31;
    const int wave = threadIdx.x >> 5;
    const int b0   = (blockIdx.x * WAVES_PER_BLOCK + wave) * ROWS_PER_WAVE;
    const int l15  = lane & 15;      // A: batch row of tile / D: hidden column
    const int half = lane >> 4;      // 0 => K 0..7, 1 => K 8..15 of A

    // ---- B matrix (constant over t): W1 padded to 16x16, bias b1 at row K=16.
    // 16-bit B 32x16 layout: lanes 0-15: N=lane,    elems 0..7 = K 0..7,  elem 8 = K16
    //                        lanes16-31: N=lane-16, elems 0..7 = K 8..15, elems 8..15 = K24..31
    v16h bm;
    #pragma unroll
    for (int e = 0; e < 16; ++e)
        bm[e] = (l15 < H_SZ && e < 8) ? (_Float16)W1[(e + 8 * half) * H_SZ + l15] : (_Float16)0.0f;
    if (half == 0) bm[8] = (l15 < H_SZ) ? (_Float16)b1[l15] : (_Float16)0.0f;  // K==16 row

    // ---- A matrix skeleton. 16-bit A 16x32 layout:
    // lanes 0-15 : row M=lane,   elems 0..7 = K 0..7,  elem 8 = K16 (bias one)
    // lanes 16-31: row M=lane-16, elems 0..7 = K 8..15, elems 8..15 = K24..31 (=0)
    v16h am;
    #pragma unroll
    for (int e = 0; e < 16; ++e) am[e] = (_Float16)0.0f;
    if (half == 0) am[8] = (_Float16)1.0f;   // multiplies the b1 row every step

    // hidden-state tile lives in the C/D accumulator layout for the whole scan
    v8f h = {0.f, 0.f, 0.f, 0.f, 0.f, 0.f, 0.f, 0.f};

    // lane's 8-float slice of x[b0+l15, t, :]
    const float* xr = x + ((size_t)(b0 + l15) * SEQ) * IN_F + half * 8;

    #pragma unroll 4
    for (int t = 0; t < SEQ; ++t) {
        const v4f p = __builtin_nontemporal_load((const v4f*)(xr + t * IN_F));
        const v4f q = __builtin_nontemporal_load((const v4f*)(xr + t * IN_F + 4));
        am[0] = (_Float16)p.x;  am[1] = (_Float16)p.y;
        am[2] = (_Float16)p.z;  am[3] = (_Float16)p.w;
        am[4] = (_Float16)q.x;  am[5] = (_Float16)q.y;
        am[6] = (_Float16)q.z;  am[7] = (_Float16)q.w;

        // D = x_t @ W1 + b1 + h   (bias folded in via the K=16 augmentation)
        h = __builtin_amdgcn_wmma_f32_16x16x32_f16(
                /*neg_a=*/false, am, /*neg_b=*/false, bm,
                /*c_mod=*/(short)0, h, /*reuse_a=*/false, /*reuse_b=*/false);

        #pragma unroll
        for (int r = 0; r < 8; ++r) h[r] = fast_tanh(h[r]);
        // padded cols 10..15: tanh(0 + 0) == 0 forever, no contamination
    }

    // ---- readout: sigmoid(h @ W2 + b2). Transpose tile through LDS. ----
    __shared__ float hs[WAVES_PER_BLOCK][16][17];   // stride 17: no bank conflicts
    #pragma unroll
    for (int r = 0; r < 8; ++r)
        hs[wave][r + 8 * half][l15] = h[r];         // [M][N] = D element
    __syncthreads();

    if (lane < 16) {
        const int m = lane;                          // batch row of the tile
        float acc[OUT_F];
        #pragma unroll
        for (int j = 0; j < OUT_F; ++j) acc[j] = b2[j];
        #pragma unroll
        for (int n = 0; n < H_SZ; ++n) {
            const float hv = hs[wave][m][n];
            #pragma unroll
            for (int j = 0; j < OUT_F; ++j)
                acc[j] = __builtin_fmaf(hv, W2[n * OUT_F + j], acc[j]);
        }
        float* o = out + (size_t)(b0 + m) * OUT_F;
        #pragma unroll
        for (int j = 0; j < OUT_F; ++j) o[j] = fast_sigmoid(acc[j]);
    }
}

extern "C" void kernel_launch(void* const* d_in, const int* in_sizes, int n_in,
                              void* d_out, int out_size, void* d_ws, size_t ws_size,
                              hipStream_t stream) {
    const float* x  = (const float*)d_in[0];
    const float* W1 = (const float*)d_in[1];
    const float* b1 = (const float*)d_in[2];
    const float* W2 = (const float*)d_in[3];
    const float* b2 = (const float*)d_in[4];
    float* out = (float*)d_out;

    const int blocks = BATCH / (WAVES_PER_BLOCK * ROWS_PER_WAVE);  // 512
    rnn_scan_wmma<<<blocks, WAVES_PER_BLOCK * 32, 0, stream>>>(x, W1, b1, W2, b2, out);
}
